// RNNFast_31421980738257
// MI455X (gfx1250) — matrix operand order repaired and checked by reading, not compiled, not run on hardware
//
#include <hip/hip_runtime.h>
#include <hip/hip_bf16.h>

// ---------------------------------------------------------------------------
// CDNA5 (gfx1250) fused 2-layer LSTM + linear head.
//   * bf16 WMMA (v_wmma_f32_16x16x32_bf16) for all matmuls, f32 accumulation
//   * recurrence: 4 persistent workgroups (16 batch rows each, batch rows are
//     independent) -> only a per-WGP s_barrier per timestep, no global sync
//   * h state double-buffered in LDS; weights bf16 in d_ws, L2-resident
//   * accumulators start from constant-zero C (inline-0 WMMA operand); bias
//     folded into the tanh-form gate epilogue (native V_TANH_F32 trans op)
//   * x stream prefetched ahead with global_prefetch_b8
//   * workspace use: bf16 weights (~1.7MB) + y0,y1 bf16 (128MB each)
// ---------------------------------------------------------------------------

typedef __bf16 bf16_t;
typedef __attribute__((ext_vector_type(16))) __bf16 v16bf;
typedef __attribute__((ext_vector_type(8)))  __bf16 v8bf;
typedef __attribute__((ext_vector_type(8)))  float  v8f;

#define B_   64
#define T_   4096
#define IN_  64
#define H_   256
#define OUT_ 64
#define G4H  (4 * H_)          // 1024 gate rows
#define HPAD (H_ + 8)          // LDS row padding (bank spread, keeps 16B align)

// Native CDNA5 transcendental tanh if the toolchain exposes it.
#if __has_builtin(__builtin_amdgcn_tanhf)
__device__ __forceinline__ float fast_tanh(float x) { return __builtin_amdgcn_tanhf(x); }
#else
__device__ __forceinline__ float fast_tanh(float x) { return tanhf(x); }
#endif

// ---- A fragment (16x32, 16-bit): lanes 0-15 -> K {0..7,16..23},
//      lanes 16-31 -> K {8..15,24..31}; element e in v16bf = vgpr e/2, half e%2.
__device__ __forceinline__ v16bf load_a_frag_bf16(const bf16_t* rowp, int kh) {
  v8bf lo = *(const v8bf*)(rowp + 8 * kh);        // k = 8*kh .. 8*kh+7
  v8bf hi = *(const v8bf*)(rowp + 16 + 8 * kh);   // k = 16+8*kh .. +7
  v16bf a;
#pragma unroll
  for (int e = 0; e < 8; ++e) { a[e] = lo[e]; a[e + 8] = hi[e]; }
  return a;
}
__device__ __forceinline__ v16bf load_a_frag_f32(const float* rowp, int kh) {
  v16bf a;
#pragma unroll
  for (int e = 0; e < 8; ++e) {
    a[e]     = (bf16_t)rowp[8 * kh + e];
    a[e + 8] = (bf16_t)rowp[16 + 8 * kh + e];
  }
  return a;
}
// ---- B fragment (32x16, 16-bit): B(k,n) = W[nbase+n][kbase+k];
//      lanes 0-15 -> K {0..7,16..23}, lanes 16-31 -> K {8..15,24..31} (same as A).
__device__ __forceinline__ v16bf load_b_frag(const bf16_t* __restrict__ w,
                                             int ldk, int nbase, int kbase,
                                             int lane) {
  const int n = lane & 15, kh = lane >> 4;
  const bf16_t* rowp = w + (size_t)(nbase + n) * ldk + kbase + 8 * kh;
  v8bf lo = *(const v8bf*)rowp;
  v8bf hi = *(const v8bf*)(rowp + 16);
  v16bf b;
#pragma unroll
  for (int e = 0; e < 8; ++e) { b[e] = lo[e]; b[e + 8] = hi[e]; }
  return b;
}

// ---------------------------------------------------------------------------
// fp32 -> bf16 weight conversion
// ---------------------------------------------------------------------------
__global__ void cvt_f32_to_bf16(const float* __restrict__ s,
                                bf16_t* __restrict__ d, int n) {
  int i = blockIdx.x * blockDim.x + threadIdx.x;
  if (i < n) d[i] = (bf16_t)s[i];
}

// ---------------------------------------------------------------------------
// Persistent LSTM layer scan. grid = B/16 workgroups, 512 threads (16 waves).
// Wave w owns hidden columns [16w,16w+16) and its matching i/f/g/o tiles.
// ---------------------------------------------------------------------------
template <int DIN, bool XF32>
__global__ __launch_bounds__(512) void lstm_scan_kernel(
    const void* __restrict__ x_v,          // [B,T,DIN] f32 (layer0) / bf16
    const bf16_t* __restrict__ wih,        // [4H,DIN] bf16
    const bf16_t* __restrict__ whh,        // [4H,H] bf16
    const float*  __restrict__ bias,       // [4H] (b_ih + b_hh)
    bf16_t* __restrict__ y,                // [B,T,H] bf16 (layer output)
    float* __restrict__ h_fin,             // [B,H] final h
    float* __restrict__ c_fin)             // [B,H] final c
{
  __shared__ bf16_t hbuf[2][16][HPAD];

  const int tid  = threadIdx.x;
  const int lane = tid & 31;
  const int wave = tid >> 5;             // 0..15
  const int kh   = lane >> 4;            // K-half / M-half selector
  const int nl   = lane & 15;            // column-in-tile / row-in-tile
  const int m0   = blockIdx.x * 16;      // batch block
  const int nc   = wave * 16;            // hidden column block of this wave

  // zero h_0
  for (int i = tid; i < 16 * HPAD; i += 512) ((bf16_t*)hbuf[0])[i] = (bf16_t)0.0f;
  __syncthreads();

  // Bias folded into the gate epilogue:
  //   sigmoid(x+b) = 0.5 + 0.5*tanh(0.5*x + 0.5*b)   (gates i, f, o)
  //   tanh(x+b)                                       (gate g)
  float hbias[4];
#pragma unroll
  for (int g = 0; g < 4; ++g) hbias[g] = 0.5f * bias[g * H_ + nc + nl];
  const float gbias = bias[2 * H_ + nc + nl];

  v8f c_st = {};                          // c tile, C layout: M = r + 8*kh
  float hv[8];
#pragma unroll
  for (int r = 0; r < 8; ++r) hv[r] = 0.0f;

  const int    mrow = m0 + nl;            // A-matrix row this lane supplies
  const float*  xf  = (const float*)x_v;
  const bf16_t* xb  = (const bf16_t*)x_v;

#pragma unroll 1
  for (int t = 0; t < T_; ++t) {
    const int cur = t & 1, nxt = cur ^ 1;

    v8f acc[4] = {};                      // constant-zero C -> inline-0 WMMA

    // prefetch the x row 8 timesteps ahead (global_prefetch_b8)
    {
      const int tp = (t + 8 < T_) ? (t + 8) : t;
      if (XF32) __builtin_prefetch(xf + ((size_t)mrow * T_ + tp) * DIN, 0, 1);
      else      __builtin_prefetch(xb + ((size_t)mrow * T_ + tp) * DIN, 0, 1);
    }

    // ---- input contribution: x_t @ w_ih^T (weights stream from L2)
#pragma unroll
    for (int kt = 0; kt < DIN / 32; ++kt) {
      v16bf afrag;
      if (XF32)
        afrag = load_a_frag_f32(xf + ((size_t)mrow * T_ + t) * DIN + kt * 32, kh);
      else
        afrag = load_a_frag_bf16(xb + ((size_t)mrow * T_ + t) * DIN + kt * 32, kh);
#pragma unroll
      for (int g = 0; g < 4; ++g) {
        v16bf bfrag = load_b_frag(wih, DIN, g * H_ + nc, kt * 32, lane);
        acc[g] = __builtin_amdgcn_wmma_f32_16x16x32_bf16(
            false, afrag, false, bfrag, (short)0, acc[g], false, false);
      }
    }

    // ---- recurrent contribution: h_{t-1} @ w_hh^T (A from LDS)
#pragma unroll
    for (int kt = 0; kt < H_ / 32; ++kt) {
      v16bf afrag = load_a_frag_bf16(&hbuf[cur][nl][kt * 32], kh);
#pragma unroll
      for (int g = 0; g < 4; ++g) {
        v16bf bfrag = load_b_frag(whh, H_, g * H_ + nc, kt * 32, lane);
        acc[g] = __builtin_amdgcn_wmma_f32_16x16x32_bf16(
            false, afrag, false, bfrag, (short)0, acc[g], false, false);
      }
    }

    // ---- gate nonlinearities + state update (wave-local: i/f/g/o co-owned)
#pragma unroll
    for (int r = 0; r < 8; ++r) {
      float iv = fmaf(0.5f, fast_tanh(fmaf(0.5f, acc[0][r], hbias[0])), 0.5f);
      float fv = fmaf(0.5f, fast_tanh(fmaf(0.5f, acc[1][r], hbias[1])), 0.5f);
      float gv = fast_tanh(acc[2][r] + gbias);
      float ov = fmaf(0.5f, fast_tanh(fmaf(0.5f, acc[3][r], hbias[3])), 0.5f);
      float cv = fmaf(fv, c_st[r], iv * gv);
      c_st[r] = cv;
      hv[r]   = ov * fast_tanh(cv);
    }

    // ---- publish h_t: LDS (double buffer) + y stream to global
#pragma unroll
    for (int r = 0; r < 8; ++r) {
      const int M  = r + 8 * kh;          // C-layout row
      bf16_t hb = (bf16_t)hv[r];
      hbuf[nxt][M][nc + nl] = hb;
      y[((size_t)(m0 + M) * T_ + t) * H_ + nc + nl] = hb;
    }
    if (t == T_ - 1) {                    // uniform branch
#pragma unroll
      for (int r = 0; r < 8; ++r) {
        const int M = r + 8 * kh;
        h_fin[(size_t)(m0 + M) * H_ + nc + nl] = hv[r];
        c_fin[(size_t)(m0 + M) * H_ + nc + nl] = c_st[r];
      }
    }
    __syncthreads();                      // one WGP-local barrier per step
  }
}

// ---------------------------------------------------------------------------
// Head GEMM: out[B*T,64] = y1[B*T,256] @ w_lin^T + b. 8 waves/block, one
// 16-row M-tile per wave, 4 N-tiles each, K=256 (8 bf16 WMMAs per N-tile).
// ---------------------------------------------------------------------------
__global__ __launch_bounds__(256) void head_gemm_kernel(
    const bf16_t* __restrict__ y1, const bf16_t* __restrict__ wlin,
    const float* __restrict__ blin, float* __restrict__ out) {
  const int lane = threadIdx.x & 31;
  const int wave = threadIdx.x >> 5;      // 0..7
  const int kh = lane >> 4, nl = lane & 15;
  const size_t row0 = ((size_t)blockIdx.x * 8 + wave) * 16;
  const bf16_t* arow = y1 + (row0 + nl) * H_;

  v8f acc[4] = {};                        // inline-0 C; bias added at store
#pragma unroll
  for (int kt = 0; kt < H_ / 32; ++kt) {
    v16bf afrag = load_a_frag_bf16(arow + kt * 32, kh);
#pragma unroll
    for (int nt = 0; nt < 4; ++nt) {
      v16bf bfrag = load_b_frag(wlin, H_, nt * 16, kt * 32, lane);
      acc[nt] = __builtin_amdgcn_wmma_f32_16x16x32_bf16(
          false, afrag, false, bfrag, (short)0, acc[nt], false, false);
    }
  }
#pragma unroll
  for (int nt = 0; nt < 4; ++nt) {
    const float bv = blin[nt * 16 + nl];
#pragma unroll
    for (int r = 0; r < 8; ++r) {
      out[(row0 + r + 8 * kh) * OUT_ + nt * 16 + nl] = acc[nt][r] + bv;
    }
  }
}

// ---------------------------------------------------------------------------
extern "C" void kernel_launch(void* const* d_in, const int* in_sizes, int n_in,
                              void* d_out, int out_size, void* d_ws,
                              size_t ws_size, hipStream_t stream) {
  (void)in_sizes; (void)n_in; (void)out_size; (void)ws_size;
  const float* x    = (const float*)d_in[0];  // [B,T,IN]
  const float* wih0 = (const float*)d_in[1];  // [4H,IN]
  const float* whh0 = (const float*)d_in[2];  // [4H,H]
  const float* b0   = (const float*)d_in[3];  // [4H]
  const float* wih1 = (const float*)d_in[4];  // [4H,H]
  const float* whh1 = (const float*)d_in[5];  // [4H,H]
  const float* b1   = (const float*)d_in[6];  // [4H]
  const float* wlin = (const float*)d_in[7];  // [OUT,H]
  const float* blin = (const float*)d_in[8];  // [OUT]

  // workspace carve-out (~258 MB): bf16 weights + y0/y1 activations
  char* ws = (char*)d_ws;
  size_t off = 0;
  auto carve = [&](size_t bytes) -> void* {
    void* p = ws + off;
    off = (off + bytes + 255) & ~(size_t)255;
    return p;
  };
  bf16_t* wih0b = (bf16_t*)carve((size_t)G4H * IN_ * 2);
  bf16_t* whh0b = (bf16_t*)carve((size_t)G4H * H_ * 2);
  bf16_t* wih1b = (bf16_t*)carve((size_t)G4H * H_ * 2);
  bf16_t* whh1b = (bf16_t*)carve((size_t)G4H * H_ * 2);
  bf16_t* wlinb = (bf16_t*)carve((size_t)OUT_ * H_ * 2);
  bf16_t* y0    = (bf16_t*)carve((size_t)B_ * T_ * H_ * 2);
  bf16_t* y1    = (bf16_t*)carve((size_t)B_ * T_ * H_ * 2);

  float* out  = (float*)d_out;                       // [B,T,OUT]
  float* hout = out + (size_t)B_ * T_ * OUT_;        // [2,B,H]
  float* cout = hout + 2 * (size_t)B_ * H_;          // [2,B,H]

  // weight conversions
  auto cvt = [&](const float* s, bf16_t* d, int n) {
    cvt_f32_to_bf16<<<(n + 255) / 256, 256, 0, stream>>>(s, d, n);
  };
  cvt(wih0, wih0b, G4H * IN_);
  cvt(whh0, whh0b, G4H * H_);
  cvt(wih1, wih1b, G4H * H_);
  cvt(whh1, whh1b, G4H * H_);
  cvt(wlin, wlinb, OUT_ * H_);

  // layer 0 scan (fp32 input), layer 1 scan (bf16 input), head GEMM
  lstm_scan_kernel<IN_, true><<<B_ / 16, 512, 0, stream>>>(
      (const void*)x, wih0b, whh0b, b0, y0, hout, cout);
  lstm_scan_kernel<H_, false><<<B_ / 16, 512, 0, stream>>>(
      (const void*)y0, wih1b, whh1b, b1, y1,
      hout + (size_t)B_ * H_, cout + (size_t)B_ * H_);
  head_gemm_kernel<<<(B_ * T_ / 16) / 8, 256, 0, stream>>>(y1, wlinb, blin, out);
}
